// AttFlowLayer_1305670058194
// MI455X (gfx1250) — compile-verified, matrix operand order checked by reading, not run on hardware
//
#include <hip/hip_runtime.h>
#include <stdint.h>

#define B_ 32
#define T_ 1024
#define J_ 256
#define D_ 512

typedef __attribute__((ext_vector_type(16))) __bf16 v16bf;
typedef __attribute__((ext_vector_type(8)))  float  v8f;

__device__ __forceinline__ uint16_t f2bf(float x) {
  uint32_t u = __float_as_uint(x);
  u += 0x7FFFu + ((u >> 16) & 1u);      // round-to-nearest-even
  return (uint16_t)(u >> 16);
}

union FragU { v16bf v; uint4 q[2]; };

// Load a 16x32 bf16 WMMA fragment (A layout; B uses the symmetric layout with
// lane = column, which for "striped dimension is row-major-contiguous" storage
// is the exact same per-lane contiguous pattern).
__device__ __forceinline__ v16bf load_frag16(const uint16_t* base, int stride, int lane) {
  int m  = lane & 15;
  int kh = (lane >> 4) << 3;            // 0 or 8
  const uint16_t* p = base + (size_t)m * stride + kh;
  FragU f;
  f.q[0] = *(const uint4*)(p);          // K = kh .. kh+7
  f.q[1] = *(const uint4*)(p + 16);     // K = kh+16 .. kh+23
  return f.v;
}

// Load one A fragment + four B fragments for k-chunk kk.
__device__ __forceinline__ void load_group(const uint16_t* __restrict__ Ab,
                                           const uint16_t* __restrict__ Bb,
                                           int stride, int kk, int lane,
                                           v16bf& a, v16bf b[4]) {
  a = load_frag16(Ab + kk, stride, lane);
#pragma unroll
  for (int s = 0; s < 4; ++s)
    b[s] = load_frag16(Bb + (size_t)(s * 16) * stride + kk, stride, lane);
}

__device__ __forceinline__ void mma4(const v16bf& a, const v16bf b[4], v8f acc[4]) {
#pragma unroll
  for (int s = 0; s < 4; ++s)
    acc[s] = __builtin_amdgcn_wmma_f32_16x16x32_bf16(false, a, false, b[s],
                                                     (short)0, acc[s], false, false);
}

// ---- prep: c1 = ctx@w1, Abf = bf16(ctx*w3) -------------------------------
__global__ void prep_context(const float* __restrict__ ctx, const float* __restrict__ w,
                             uint16_t* __restrict__ Abf, float* __restrict__ c1) {
  __shared__ float red[128];
  int row = blockIdx.x;                 // b*T + t
  int tid = threadIdx.x;                // 128
  const float* w1 = w;
  const float* w3 = w + 2 * D_;
  const float* cr = ctx + (size_t)row * D_;
  float s = 0.f;
  for (int d = tid; d < D_; d += 128) {
    float c = cr[d];
    s += c * w1[d];
    Abf[(size_t)row * D_ + d] = f2bf(c * w3[d]);
  }
  red[tid] = s; __syncthreads();
  for (int k = 64; k > 0; k >>= 1) { if (tid < k) red[tid] += red[tid + k]; __syncthreads(); }
  if (tid == 0) c1[row] = red[0];
}

// ---- prep: q2 = qry@w2, Qbf = bf16(qry), QTbf = bf16(qry)^T --------------
__global__ void prep_query(const float* __restrict__ qry, const float* __restrict__ w,
                           uint16_t* __restrict__ Qbf, uint16_t* __restrict__ QTbf,
                           float* __restrict__ q2) {
  __shared__ float red[128];
  int row = blockIdx.x;                 // b*J + j
  int b = row >> 8;
  int j = row & (J_ - 1);
  int tid = threadIdx.x;
  const float* w2 = w + D_;
  const float* qr = qry + (size_t)row * D_;
  float s = 0.f;
  for (int d = tid; d < D_; d += 128) {
    float q = qr[d];
    s += q * w2[d];
    uint16_t bv = f2bf(q);
    Qbf[(size_t)row * D_ + d] = bv;
    QTbf[((size_t)(b * D_ + d)) * J_ + j] = bv;
  }
  red[tid] = s; __syncthreads();
  for (int k = 64; k > 0; k >>= 1) { if (tid < k) red[tid] += red[tid + k]; __syncthreads(); }
  if (tid == 0) q2[row] = red[0];
}

// ---- GEMM 1: S[b,t,j] = (ctx*w3)@Q^T + c1[t] + q2[j] ---------------------
// Ping-pong double buffering, unrolled by 2 k-chunks: no buffer-rotation
// register copies; loads for k+64 issue right after the WMMAs consuming the
// same buffer, keeping ~4 WMMAs between a load and its consumer.
__global__ void gemm_S(const uint16_t* __restrict__ Abf, const uint16_t* __restrict__ Qbf,
                       const float* __restrict__ c1, const float* __restrict__ q2,
                       float* __restrict__ S) {
  int lane = threadIdx.x & 31;
  int wave = blockIdx.x * (blockDim.x >> 5) + (threadIdx.x >> 5);
  const int TT = T_ / 16, JT = J_ / 64;
  int b   = wave / (TT * JT);
  int rem = wave % (TT * JT);
  int t0  = (rem / JT) * 16;
  int j0  = (rem % JT) * 64;
  const uint16_t* Ab = Abf + ((size_t)(b * T_ + t0)) * D_;
  const uint16_t* Bb = Qbf + ((size_t)(b * J_ + j0)) * D_;
  v8f zero = {0.f,0.f,0.f,0.f,0.f,0.f,0.f,0.f};
  v8f acc[4]; for (int s = 0; s < 4; ++s) acc[s] = zero;

  v16bf a0, a1, b0[4], b1[4];
  load_group(Ab, Bb, D_, 0,  lane, a0, b0);
  load_group(Ab, Bb, D_, 32, lane, a1, b1);

  for (int kk = 0; kk < D_ - 64; kk += 64) {
    mma4(a0, b0, acc);
    load_group(Ab, Bb, D_, kk + 64, lane, a0, b0);
    mma4(a1, b1, acc);
    load_group(Ab, Bb, D_, kk + 96, lane, a1, b1);
  }
  mma4(a0, b0, acc);
  mma4(a1, b1, acc);

  int mh = (lane >> 4) << 3;
  int n  = lane & 15;
#pragma unroll
  for (int s = 0; s < 4; ++s)
#pragma unroll
    for (int r = 0; r < 8; ++r) {
      int t = t0 + r + mh;
      int j = j0 + s * 16 + n;
      S[((size_t)(b * T_ + t)) * J_ + j] = acc[s][r] + c1[b * T_ + t] + q2[b * J_ + j];
    }
}

// ---- column softmax stats over T (online max/sum) ------------------------
__global__ void col_stats(const float* __restrict__ S, float* __restrict__ mcol,
                          float* __restrict__ rz) {
  __shared__ float ms[256], ss[256];
  int bid = blockIdx.x;                 // B * (J/64)
  int b  = bid >> 2;
  int j0 = (bid & 3) * 64;
  int tid = threadIdx.x;
  int jj = tid & 63, tt = tid >> 6;
  int j = j0 + jj;
  const float* sb = S + (size_t)b * T_ * J_ + j;
  float mm = -INFINITY, sum = 0.f;
  for (int t = tt; t < T_; t += 4) {
    float x = sb[(size_t)t * J_];
    if (x > mm) { sum = sum * __expf(mm - x) + 1.f; mm = x; }
    else        { sum += __expf(x - mm); }
  }
  ms[tid] = mm; ss[tid] = sum; __syncthreads();
  if (tt == 0) {
    for (int k = 1; k < 4; ++k) {
      float m2 = ms[jj + 64 * k], s2 = ss[jj + 64 * k];
      if (m2 > mm) { sum = sum * __expf(mm - m2) + s2; mm = m2; }
      else         { sum += s2 * __expf(m2 - mm); }
    }
    mcol[b * J_ + j] = mm;
    rz[b * J_ + j]   = 1.f / sum;
  }
}

// ---- row max over J (one wave per row) -----------------------------------
__global__ void row_max(const float* __restrict__ S, float* __restrict__ rmax) {
  int lane = threadIdx.x & 31;
  int row  = blockIdx.x * (blockDim.x >> 5) + (threadIdx.x >> 5);  // b*T + t
  const float* sr = S + (size_t)row * J_;
  float m = -INFINITY;
  for (int j = lane; j < J_; j += 32) m = fmaxf(m, sr[j]);
  for (int i = 16; i > 0; i >>= 1) m = fmaxf(m, __shfl_xor(m, i, 32));
  if (lane == 0) rmax[row] = m;
}

// ---- b_t = softmax over T of rmax ----------------------------------------
__global__ void bt_softmax(const float* __restrict__ rmax, float* __restrict__ bt) {
  __shared__ float red[1024];
  int b = blockIdx.x, tid = threadIdx.x;
  float x = rmax[b * T_ + tid];
  red[tid] = x; __syncthreads();
  for (int k = 512; k > 0; k >>= 1) { if (tid < k) red[tid] = fmaxf(red[tid], red[tid + k]); __syncthreads(); }
  float m = red[0]; __syncthreads();
  float e = __expf(x - m);
  red[tid] = e; __syncthreads();
  for (int k = 512; k > 0; k >>= 1) { if (tid < k) red[tid] += red[tid + k]; __syncthreads(); }
  bt[b * T_ + tid] = e / red[0];
}

// ---- h[b,d] = sum_t bt[t]*ctx[b,t,d] -------------------------------------
__global__ void compute_h(const float* __restrict__ bt, const float* __restrict__ ctx,
                          float* __restrict__ h) {
  __shared__ float bts[1024];
  int b = blockIdx.x, tid = threadIdx.x;   // 512 threads
  bts[tid]       = bt[b * T_ + tid];
  bts[tid + 512] = bt[b * T_ + tid + 512];
  __syncthreads();
  float acc = 0.f;
  const float* cb = ctx + (size_t)b * T_ * D_ + tid;
  for (int t = 0; t < T_; ++t) acc += bts[t] * cb[(size_t)t * D_];
  h[b * D_ + tid] = acc;
}

// ---- P = bf16(exp(S - mcol) * invZ) --------------------------------------
__global__ void make_P(const float* __restrict__ S, const float* __restrict__ mcol,
                       const float* __restrict__ rz, uint16_t* __restrict__ Pbf) {
  size_t idx = (size_t)blockIdx.x * blockDim.x + threadIdx.x;   // b*T*J + t*J + j
  int j = (int)(idx & (J_ - 1));
  int b = (int)(idx / ((size_t)T_ * J_));
  float p = __expf(S[idx] - mcol[b * J_ + j]) * rz[b * J_ + j];
  Pbf[idx] = f2bf(p);
}

// ---- GEMM 2 + epilogue: U = P@Q ; G = [c, U, c*U, c*h] -------------------
__global__ void gemm_U_out(const uint16_t* __restrict__ Pbf, const uint16_t* __restrict__ QTbf,
                           const float* __restrict__ ctx, const float* __restrict__ h,
                           float* __restrict__ G) {
  int lane = threadIdx.x & 31;
  int wave = blockIdx.x * (blockDim.x >> 5) + (threadIdx.x >> 5);
  const int TT = T_ / 16, DT = D_ / 64;
  int b   = wave / (TT * DT);
  int rem = wave % (TT * DT);
  int t0  = (rem / DT) * 16;
  int d0  = (rem % DT) * 64;
  const uint16_t* Pb = Pbf  + ((size_t)(b * T_ + t0)) * J_;
  const uint16_t* Qb = QTbf + ((size_t)(b * D_ + d0)) * J_;
  v8f zero = {0.f,0.f,0.f,0.f,0.f,0.f,0.f,0.f};
  v8f acc[4]; for (int s = 0; s < 4; ++s) acc[s] = zero;

  v16bf a0, a1, b0[4], b1[4];
  load_group(Pb, Qb, J_, 0,  lane, a0, b0);
  load_group(Pb, Qb, J_, 32, lane, a1, b1);

  for (int kk = 0; kk < J_ - 64; kk += 64) {
    mma4(a0, b0, acc);
    load_group(Pb, Qb, J_, kk + 64, lane, a0, b0);
    mma4(a1, b1, acc);
    load_group(Pb, Qb, J_, kk + 96, lane, a1, b1);
  }
  mma4(a0, b0, acc);
  mma4(a1, b1, acc);

  int mh = (lane >> 4) << 3;
  int n  = lane & 15;
#pragma unroll
  for (int s = 0; s < 4; ++s)
#pragma unroll
    for (int r = 0; r < 8; ++r) {
      int t = t0 + r + mh;
      int d = d0 + s * 16 + n;
      float Uv = acc[s][r];
      float c  = ctx[((size_t)(b * T_ + t)) * D_ + d];
      float hv = h[b * D_ + d];
      float* gr = G + ((size_t)(b * T_ + t)) * (4 * D_);
      gr[d]          = c;
      gr[D_ + d]     = Uv;
      gr[2 * D_ + d] = c * Uv;
      gr[3 * D_ + d] = c * hv;
    }
}

extern "C" void kernel_launch(void* const* d_in, const int* in_sizes, int n_in,
                              void* d_out, int out_size, void* d_ws, size_t ws_size,
                              hipStream_t stream) {
  (void)in_sizes; (void)n_in; (void)out_size; (void)ws_size;
  const float* ctx = (const float*)d_in[0];
  const float* qry = (const float*)d_in[1];
  const float* w   = (const float*)d_in[2];
  float* G = (float*)d_out;

  char* ws = (char*)d_ws;
  size_t off = 0;
  auto carve = [&](size_t bytes) { char* p = ws + off; off += (bytes + 255) & ~(size_t)255; return p; };
  uint16_t* Abf  = (uint16_t*)carve((size_t)B_ * T_ * D_ * 2);   // bf16 ctx*w3
  uint16_t* Qbf  = (uint16_t*)carve((size_t)B_ * J_ * D_ * 2);   // bf16 query
  uint16_t* QTbf = (uint16_t*)carve((size_t)B_ * D_ * J_ * 2);   // bf16 query^T
  uint16_t* Pbf  = (uint16_t*)carve((size_t)B_ * T_ * J_ * 2);   // bf16 softmax probs
  float* S    = (float*)carve((size_t)B_ * T_ * J_ * 4);
  float* c1   = (float*)carve((size_t)B_ * T_ * 4);
  float* q2   = (float*)carve((size_t)B_ * J_ * 4);
  float* mcol = (float*)carve((size_t)B_ * J_ * 4);
  float* rz   = (float*)carve((size_t)B_ * J_ * 4);
  float* rmax = (float*)carve((size_t)B_ * T_ * 4);
  float* bt   = (float*)carve((size_t)B_ * T_ * 4);
  float* h    = (float*)carve((size_t)B_ * D_ * 4);

  prep_context<<<B_ * T_, 128, 0, stream>>>(ctx, w, Abf, c1);
  prep_query  <<<B_ * J_, 128, 0, stream>>>(qry, w, Qbf, QTbf, q2);
  gemm_S      <<<(B_ * (T_ / 16) * (J_ / 64)) / 8, 256, 0, stream>>>(Abf, Qbf, c1, q2, S);
  col_stats   <<<B_ * (J_ / 64), 256, 0, stream>>>(S, mcol, rz);
  row_max     <<<(B_ * T_) / 8, 256, 0, stream>>>(S, rmax);
  bt_softmax  <<<B_, 1024, 0, stream>>>(rmax, bt);
  compute_h   <<<B_, 512, 0, stream>>>(bt, ctx, h);
  make_P      <<<(B_ * T_ * J_) / 256, 256, 0, stream>>>(S, mcol, rz, Pbf);
  gemm_U_out  <<<(B_ * (T_ / 16) * (D_ / 64)) / 8, 256, 0, stream>>>(Pbf, QTbf, ctx, h, G);
}
